// MoE_609885356951
// MI455X (gfx1250) — matrix operand order repaired and checked by reading, not compiled, not run on hardware
//
#include <hip/hip_runtime.h>
#include <cstdint>
#include <cstddef>

// ---------------- problem constants ----------------
#define T_TOK   8192      // BATCH*SEQ
#define DMODEL  1024
#define HIDDEN_ 4096
#define RHID    1024
#define NE      8

typedef _Float16 f16;
typedef __attribute__((ext_vector_type(4)))  _Float16 v4h;
typedef __attribute__((ext_vector_type(8)))  _Float16 v8h;
typedef __attribute__((ext_vector_type(16))) _Float16 v16h;
typedef __attribute__((ext_vector_type(8)))  float    v8f;
typedef __attribute__((ext_vector_type(4)))  int      v4i;

// ---------------- CDNA5 async global->LDS path ----------------
#if defined(__has_builtin)
#if __has_builtin(__builtin_amdgcn_global_load_async_to_lds_b128)
#define ASYNC_LDS 1
#endif
#endif

__device__ __forceinline__ void cp_b128(const f16* g, f16* l) {
#ifdef ASYNC_LDS
  // signature: (int4 AS1* gsrc, int4 AS3* ldst, imm offset, imm cpol)
  __builtin_amdgcn_global_load_async_to_lds_b128(
      (__attribute__((address_space(1))) v4i*)
          (__attribute__((address_space(1))) void*)(f16*)g,
      (__attribute__((address_space(3))) v4i*)
          (__attribute__((address_space(3))) void*)l,
      0, 0);
#else
  *(v8h*)l = *(const v8h*)g;
#endif
}

__device__ __forceinline__ void wait_async_all() {
#ifdef ASYNC_LDS
#if __has_builtin(__builtin_amdgcn_s_wait_asynccnt)
  __builtin_amdgcn_s_wait_asynccnt(0);
#else
  asm volatile("s_wait_asynccnt 0x0" ::: "memory");
#endif
#endif
}

// ---------------- WMMA helpers (CDNA5 16x16x32 f16 -> f32) ----------------
__device__ __forceinline__ v16h mk16(v8h lo, v8h hi) {
  v16h r;
#pragma unroll
  for (int i = 0; i < 8; ++i) { r[i] = lo[i]; r[i + 8] = hi[i]; }
  return r;
}

__device__ __forceinline__ v8f wmma16(v16h a, v16h b, v8f c) {
  // (neg_a, A, neg_b, B, c_mod, C, reuse_a, reuse_b)
  return __builtin_amdgcn_wmma_f32_16x16x32_f16(false, a, false, b, (short)0, c,
                                                false, false);
}

// LDS A tile: [64 rows][LDA=40 f16], row-major, K-slice of 32.
// lane holds row m, elements 0..7 = K g*8.., elements 8..15 = K 16+g*8..
__device__ __forceinline__ v16h frag_a(const f16* As, int m, int g) {
  const f16* q = As + m * 40 + g * 8;
  return mk16(*(const v8h*)q, *(const v8h*)(q + 16));
}
// LDS B tile: [128 rows(n)][LDB=40 f16] storing W[n][k0..k0+31].
// lane holds col n, elements 0..7 = K g*8.., elements 8..15 = K 16+g*8..
__device__ __forceinline__ v16h frag_b(const f16* Bs, int n, int g) {
  const f16* q = Bs + n * 40 + g * 8;
  return mk16(*(const v8h*)q, *(const v8h*)(q + 16));
}

__device__ __forceinline__ float silu_f(float x) {
  return x / (1.0f + __expf(-x));
}

// ---------------- f32 -> f16 convert ----------------
__global__ __launch_bounds__(256) void k_cvt(const float4* __restrict__ s,
                                             v4h* __restrict__ d, int n4) {
  int i = blockIdx.x * 256 + threadIdx.x;
  if (i < n4) {
    float4 v = s[i];
    v4h o;
    o[0] = (_Float16)v.x; o[1] = (_Float16)v.y;
    o[2] = (_Float16)v.z; o[3] = (_Float16)v.w;
    d[i] = o;
  }
}

// ---------------- router ----------------
__global__ __launch_bounds__(256) void k_init(int* __restrict__ counts,
                                              int* __restrict__ cursor) {
  if (threadIdx.x < NE) { counts[threadIdx.x] = 0; cursor[threadIdx.x] = 0; }
}

__global__ __launch_bounds__(256) void k_router(const float* __restrict__ X,
                                                const float* __restrict__ RW,
                                                const float* __restrict__ bias,
                                                int* __restrict__ counts,
                                                int* __restrict__ pe,
                                                float* __restrict__ pg) {
  const int lane = threadIdx.x & 31, wid = threadIdx.x >> 5;
  const int t = blockIdx.x * 8 + wid;           // one wave per token
  float acc[NE] = {};
  const float* xr = X + (size_t)t * DMODEL;
  for (int j = lane; j < DMODEL; j += 32) {
    float xv = xr[j];
#pragma unroll
    for (int e = 0; e < NE; ++e) acc[e] += xv * RW[e * DMODEL + j];
  }
#pragma unroll
  for (int e = 0; e < NE; ++e) {
#pragma unroll
    for (int off = 16; off > 0; off >>= 1) acc[e] += __shfl_xor(acc[e], off, 32);
  }
  if (lane == 0) {
    float s[NE], adj[NE];
#pragma unroll
    for (int e = 0; e < NE; ++e) {
      s[e] = 1.0f / (1.0f + __expf(-acc[e]));
      adj[e] = s[e] + bias[e];
    }
    int e0 = 0; float b0 = adj[0];
#pragma unroll
    for (int e = 1; e < NE; ++e) if (adj[e] > b0) { b0 = adj[e]; e0 = e; }
    int e1 = -1; float b1 = -3.4e38f;
#pragma unroll
    for (int e = 0; e < NE; ++e)
      if (e != e0 && adj[e] > b1) { b1 = adj[e]; e1 = e; }
    atomicAdd(&counts[e0], 1);
    atomicAdd(&counts[e1], 1);
    pe[t * 2 + 0] = e0; pe[t * 2 + 1] = e1;
    pg[t * 2 + 0] = s[e0]; pg[t * 2 + 1] = s[e1];
  }
}

__global__ void k_offsets(const int* __restrict__ counts, int* __restrict__ offs) {
  if (threadIdx.x == 0) {
    int a = 0;
    for (int e = 0; e < NE; ++e) { offs[e] = a; a += counts[e]; }
  }
}

__global__ __launch_bounds__(256) void k_scatter(const int* __restrict__ pe,
                                                 const float* __restrict__ pg,
                                                 const int* __restrict__ offs,
                                                 int* __restrict__ cursor,
                                                 int* __restrict__ tok,
                                                 float* __restrict__ gate) {
  int t = blockIdx.x * 256 + threadIdx.x;
  if (t < T_TOK) {
#pragma unroll
    for (int k = 0; k < 2; ++k) {
      int e = pe[t * 2 + k];
      int pos = atomicAdd(&cursor[e], 1);
      int slot = offs[e] + pos;
      tok[slot] = t;
      gate[slot] = pg[t * 2 + k];
    }
  }
}

// ---------------- GEMM 1: Hs = silu(x@W1^T) * (x@W2^T), f16 out ----------------
// block: 256 thr (8 waves, 2x4), tile 64x128, BK=32, K=DMODEL
// double-buffered LDS + async global->LDS loads
__global__ __launch_bounds__(256) void k_shared_h(const f16* __restrict__ X,
                                                  const f16* __restrict__ W1,
                                                  const f16* __restrict__ W2,
                                                  f16* __restrict__ Hs) {
  __shared__ f16 As[2][64 * 40];
  __shared__ f16 B0s[2][128 * 40];
  __shared__ f16 B1s[2][128 * 40];
  const int tid = threadIdx.x, lane = tid & 31, wid = tid >> 5;
  const int g = lane >> 4, ln = lane & 15;
  const int wM = wid >> 2, wN = wid & 3;
  const int m0 = blockIdx.x * 64, n0 = blockIdx.y * 128;

  const int ar = tid >> 2, aseg = tid & 3;
  const int dA = ar * 40 + aseg * 8;           // LDS dest offsets (f16 units)
  const int dBa = ar * 40 + aseg * 8;
  const int dBb = (ar + 64) * 40 + aseg * 8;
  const f16* aRow = X + (size_t)(m0 + ar) * DMODEL + aseg * 8;
  const f16* pB0a = W1 + (size_t)(n0 + ar) * DMODEL + aseg * 8;
  const f16* pB0b = W1 + (size_t)(n0 + ar + 64) * DMODEL + aseg * 8;
  const f16* pB1a = W2 + (size_t)(n0 + ar) * DMODEL + aseg * 8;
  const f16* pB1b = W2 + (size_t)(n0 + ar + 64) * DMODEL + aseg * 8;

  v8f acc1[2][2] = {}; v8f acc2[2][2] = {};

  const int NK = DMODEL / 32;
  // prologue: tile 0 into buffer 0
  cp_b128(aRow, &As[0][dA]);
  cp_b128(pB0a, &B0s[0][dBa]);
  cp_b128(pB0b, &B0s[0][dBb]);
  cp_b128(pB1a, &B1s[0][dBa]);
  cp_b128(pB1b, &B1s[0][dBb]);
  wait_async_all();
  __syncthreads();

  for (int kk = 0; kk < NK; ++kk) {
    const int buf = kk & 1;
    if (kk + 1 < NK) {
      const int k1 = (kk + 1) * 32, nb = buf ^ 1;
      cp_b128(aRow + k1, &As[nb][dA]);
      cp_b128(pB0a + k1, &B0s[nb][dBa]);
      cp_b128(pB0b + k1, &B0s[nb][dBb]);
      cp_b128(pB1a + k1, &B1s[nb][dBa]);
      cp_b128(pB1b + k1, &B1s[nb][dBb]);
    }
    v16h a[2];
#pragma unroll
    for (int ms = 0; ms < 2; ++ms)
      a[ms] = frag_a(As[buf], wM * 32 + ms * 16 + ln, g);
#pragma unroll
    for (int ns = 0; ns < 2; ++ns) {
      v16h b0 = frag_b(B0s[buf], wN * 32 + ns * 16 + ln, g);
      v16h b1 = frag_b(B1s[buf], wN * 32 + ns * 16 + ln, g);
#pragma unroll
      for (int ms = 0; ms < 2; ++ms) {
        acc1[ms][ns] = wmma16(a[ms], b0, acc1[ms][ns]);
        acc2[ms][ns] = wmma16(a[ms], b1, acc2[ms][ns]);
      }
    }
    if (kk + 1 < NK) {
      wait_async_all();
      __syncthreads();
    }
  }
#pragma unroll
  for (int ms = 0; ms < 2; ++ms)
#pragma unroll
    for (int ns = 0; ns < 2; ++ns) {
      int row = m0 + wM * 32 + ms * 16 + g * 8;
      int col = n0 + wN * 32 + ns * 16 + ln;
#pragma unroll
      for (int r = 0; r < 8; ++r) {
        float h1 = acc1[ms][ns][r], h2 = acc2[ms][ns][r];
        Hs[(size_t)(row + r) * HIDDEN_ + col] = (f16)(silu_f(h1) * h2);
      }
    }
}

// ---------------- GEMM 2: out = Hs @ W3^T (f32 plain store), K=HIDDEN ----------------
__global__ __launch_bounds__(256) void k_shared_out(const f16* __restrict__ Hs,
                                                    const f16* __restrict__ W3,
                                                    float* __restrict__ Out) {
  __shared__ f16 As[2][64 * 40];
  __shared__ f16 Bs[2][128 * 40];
  const int tid = threadIdx.x, lane = tid & 31, wid = tid >> 5;
  const int g = lane >> 4, ln = lane & 15;
  const int wM = wid >> 2, wN = wid & 3;
  const int m0 = blockIdx.x * 64, n0 = blockIdx.y * 128;

  const int ar = tid >> 2, aseg = tid & 3;
  const int dA = ar * 40 + aseg * 8;
  const int dBa = ar * 40 + aseg * 8;
  const int dBb = (ar + 64) * 40 + aseg * 8;
  const f16* aRow = Hs + (size_t)(m0 + ar) * HIDDEN_ + aseg * 8;
  const f16* pBa = W3 + (size_t)(n0 + ar) * HIDDEN_ + aseg * 8;
  const f16* pBb = W3 + (size_t)(n0 + ar + 64) * HIDDEN_ + aseg * 8;

  v8f acc[2][2] = {};
  const int NK = HIDDEN_ / 32;
  cp_b128(aRow, &As[0][dA]);
  cp_b128(pBa, &Bs[0][dBa]);
  cp_b128(pBb, &Bs[0][dBb]);
  wait_async_all();
  __syncthreads();

  for (int kk = 0; kk < NK; ++kk) {
    const int buf = kk & 1;
    if (kk + 1 < NK) {
      const int k1 = (kk + 1) * 32, nb = buf ^ 1;
      cp_b128(aRow + k1, &As[nb][dA]);
      cp_b128(pBa + k1, &Bs[nb][dBa]);
      cp_b128(pBb + k1, &Bs[nb][dBb]);
    }
    v16h a[2];
#pragma unroll
    for (int ms = 0; ms < 2; ++ms)
      a[ms] = frag_a(As[buf], wM * 32 + ms * 16 + ln, g);
#pragma unroll
    for (int ns = 0; ns < 2; ++ns) {
      v16h b = frag_b(Bs[buf], wN * 32 + ns * 16 + ln, g);
#pragma unroll
      for (int ms = 0; ms < 2; ++ms) acc[ms][ns] = wmma16(a[ms], b, acc[ms][ns]);
    }
    if (kk + 1 < NK) {
      wait_async_all();
      __syncthreads();
    }
  }
#pragma unroll
  for (int ms = 0; ms < 2; ++ms)
#pragma unroll
    for (int ns = 0; ns < 2; ++ns) {
      int row = m0 + wM * 32 + ms * 16 + g * 8;
      int col = n0 + wN * 32 + ns * 16 + ln;
#pragma unroll
      for (int r = 0; r < 8; ++r)
        Out[(size_t)(row + r) * DMODEL + col] = acc[ms][ns][r];
    }
}

// ---------------- GEMM 3: routed hidden, gathered A rows, dual halves of w12 ----------
// Hb[slot][n] = silu(h1)*h2*gate, grid (Mtiles, RHID/128, NE)
__global__ __launch_bounds__(256) void k_routed_h(const f16* __restrict__ X,
                                                  const f16* __restrict__ W12,
                                                  const int* __restrict__ counts,
                                                  const int* __restrict__ offs,
                                                  const int* __restrict__ tok,
                                                  const float* __restrict__ gate,
                                                  f16* __restrict__ Hb) {
  const int e = blockIdx.z;
  const int cnt = counts[e];
  const int m0 = blockIdx.x * 64;
  if (m0 >= cnt) return;
  const int base = offs[e];
  const f16* W = W12 + (size_t)e * (2 * RHID) * DMODEL;
  const int n0 = blockIdx.y * 128;

  __shared__ f16 As[2][64 * 40];
  __shared__ f16 B0s[2][128 * 40];
  __shared__ f16 B1s[2][128 * 40];
  const int tid = threadIdx.x, lane = tid & 31, wid = tid >> 5;
  const int g = lane >> 4, ln = lane & 15;
  const int wM = wid >> 2, wN = wid & 3;

  const int ar = tid >> 2, aseg = tid & 3;
  const int dA = ar * 40 + aseg * 8;
  const int dBa = ar * 40 + aseg * 8;
  const int dBb = (ar + 64) * 40 + aseg * 8;
  int mrA = m0 + ar;
  int slotA = base + (mrA < cnt ? mrA : cnt - 1);
  const f16* aRow = X + (size_t)tok[slotA] * DMODEL + aseg * 8;
  const f16* pB0a = W + (size_t)(n0 + ar) * DMODEL + aseg * 8;
  const f16* pB0b = W + (size_t)(n0 + ar + 64) * DMODEL + aseg * 8;
  const f16* pB1a = W + (size_t)(RHID + n0 + ar) * DMODEL + aseg * 8;
  const f16* pB1b = W + (size_t)(RHID + n0 + ar + 64) * DMODEL + aseg * 8;

  v8f acc1[2][2] = {}; v8f acc2[2][2] = {};
  const int NK = DMODEL / 32;
  cp_b128(aRow, &As[0][dA]);
  cp_b128(pB0a, &B0s[0][dBa]);
  cp_b128(pB0b, &B0s[0][dBb]);
  cp_b128(pB1a, &B1s[0][dBa]);
  cp_b128(pB1b, &B1s[0][dBb]);
  wait_async_all();
  __syncthreads();

  for (int kk = 0; kk < NK; ++kk) {
    const int buf = kk & 1;
    if (kk + 1 < NK) {
      const int k1 = (kk + 1) * 32, nb = buf ^ 1;
      cp_b128(aRow + k1, &As[nb][dA]);
      cp_b128(pB0a + k1, &B0s[nb][dBa]);
      cp_b128(pB0b + k1, &B0s[nb][dBb]);
      cp_b128(pB1a + k1, &B1s[nb][dBa]);
      cp_b128(pB1b + k1, &B1s[nb][dBb]);
    }
    v16h a[2];
#pragma unroll
    for (int ms = 0; ms < 2; ++ms)
      a[ms] = frag_a(As[buf], wM * 32 + ms * 16 + ln, g);
#pragma unroll
    for (int ns = 0; ns < 2; ++ns) {
      v16h b0 = frag_b(B0s[buf], wN * 32 + ns * 16 + ln, g);
      v16h b1 = frag_b(B1s[buf], wN * 32 + ns * 16 + ln, g);
#pragma unroll
      for (int ms = 0; ms < 2; ++ms) {
        acc1[ms][ns] = wmma16(a[ms], b0, acc1[ms][ns]);
        acc2[ms][ns] = wmma16(a[ms], b1, acc2[ms][ns]);
      }
    }
    if (kk + 1 < NK) {
      wait_async_all();
      __syncthreads();
    }
  }
#pragma unroll
  for (int ms = 0; ms < 2; ++ms)
#pragma unroll
    for (int ns = 0; ns < 2; ++ns) {
      int mrow = wM * 32 + ms * 16 + g * 8;
      int col = n0 + wN * 32 + ns * 16 + ln;
#pragma unroll
      for (int r = 0; r < 8; ++r) {
        int mg = m0 + mrow + r;
        if (mg < cnt) {
          int slot = base + mg;
          float gv = gate[slot];
          float h1 = acc1[ms][ns][r], h2 = acc2[ms][ns][r];
          Hb[(size_t)slot * RHID + col] = (f16)(silu_f(h1) * h2 * gv);
        }
      }
    }
}

// ---------------- GEMM 4: routed output, atomicAdd into out ----------------
__global__ __launch_bounds__(256) void k_routed_out(const f16* __restrict__ Hb,
                                                    const f16* __restrict__ W3e,
                                                    const int* __restrict__ counts,
                                                    const int* __restrict__ offs,
                                                    const int* __restrict__ tok,
                                                    float* __restrict__ Out) {
  const int e = blockIdx.z;
  const int cnt = counts[e];
  const int m0 = blockIdx.x * 64;
  if (m0 >= cnt) return;
  const int base = offs[e];
  const f16* W = W3e + (size_t)e * DMODEL * RHID;
  const int n0 = blockIdx.y * 128;

  __shared__ f16 As[2][64 * 40];
  __shared__ f16 Bs[2][128 * 40];
  const int tid = threadIdx.x, lane = tid & 31, wid = tid >> 5;
  const int g = lane >> 4, ln = lane & 15;
  const int wM = wid >> 2, wN = wid & 3;

  const int ar = tid >> 2, aseg = tid & 3;
  const int dA = ar * 40 + aseg * 8;
  const int dBa = ar * 40 + aseg * 8;
  const int dBb = (ar + 64) * 40 + aseg * 8;
  int mrA = m0 + ar;
  int slotA = base + (mrA < cnt ? mrA : cnt - 1);
  const f16* aRow = Hb + (size_t)slotA * RHID + aseg * 8;
  const f16* pBa = W + (size_t)(n0 + ar) * RHID + aseg * 8;
  const f16* pBb = W + (size_t)(n0 + ar + 64) * RHID + aseg * 8;

  v8f acc[2][2] = {};
  const int NK = RHID / 32;
  cp_b128(aRow, &As[0][dA]);
  cp_b128(pBa, &Bs[0][dBa]);
  cp_b128(pBb, &Bs[0][dBb]);
  wait_async_all();
  __syncthreads();

  for (int kk = 0; kk < NK; ++kk) {
    const int buf = kk & 1;
    if (kk + 1 < NK) {
      const int k1 = (kk + 1) * 32, nb = buf ^ 1;
      cp_b128(aRow + k1, &As[nb][dA]);
      cp_b128(pBa + k1, &Bs[nb][dBa]);
      cp_b128(pBb + k1, &Bs[nb][dBb]);
    }
    v16h a[2];
#pragma unroll
    for (int ms = 0; ms < 2; ++ms)
      a[ms] = frag_a(As[buf], wM * 32 + ms * 16 + ln, g);
#pragma unroll
    for (int ns = 0; ns < 2; ++ns) {
      v16h b = frag_b(Bs[buf], wN * 32 + ns * 16 + ln, g);
#pragma unroll
      for (int ms = 0; ms < 2; ++ms) acc[ms][ns] = wmma16(a[ms], b, acc[ms][ns]);
    }
    if (kk + 1 < NK) {
      wait_async_all();
      __syncthreads();
    }
  }
#pragma unroll
  for (int ms = 0; ms < 2; ++ms)
#pragma unroll
    for (int ns = 0; ns < 2; ++ns) {
      int mrow = wM * 32 + ms * 16 + g * 8;
      int col = n0 + wN * 32 + ns * 16 + ln;
#pragma unroll
      for (int r = 0; r < 8; ++r) {
        int mg = m0 + mrow + r;
        if (mg < cnt) {
          int t = tok[base + mg];
          atomicAdd(&Out[(size_t)t * DMODEL + col], acc[ms][ns][r]);
        }
      }
    }
}

// ---------------- host launcher ----------------
extern "C" void kernel_launch(void* const* d_in, const int* in_sizes, int n_in,
                              void* d_out, int out_size, void* d_ws, size_t ws_size,
                              hipStream_t stream) {
  (void)in_sizes; (void)n_in; (void)out_size; (void)ws_size;

  const float* x    = (const float*)d_in[0];   // (2,4096,1024)
  const float* w12  = (const float*)d_in[1];   // (8,2048,1024)
  const float* w3   = (const float*)d_in[2];   // (8,1024,1024)
  const float* sw1  = (const float*)d_in[3];   // (4096,1024)
  const float* sw2  = (const float*)d_in[4];   // (4096,1024)
  const float* sw3  = (const float*)d_in[5];   // (1024,4096)
  const float* rw   = (const float*)d_in[6];   // (8,1024)
  const float* bias = (const float*)d_in[7];   // (8,)
  float* out = (float*)d_out;

  // workspace carve (256B aligned)
  char* base = (char*)d_ws;
  size_t cur = 0;
  auto carve = [&](size_t bytes) -> void* {
    void* r = base + cur;
    cur += (bytes + 255) & ~(size_t)255;
    return r;
  };
  f16* x16   = (f16*)carve((size_t)T_TOK * DMODEL * 2);
  f16* w12h  = (f16*)carve((size_t)NE * 2 * RHID * DMODEL * 2);
  f16* w3h   = (f16*)carve((size_t)NE * DMODEL * RHID * 2);
  f16* s1h   = (f16*)carve((size_t)HIDDEN_ * DMODEL * 2);
  f16* s2h   = (f16*)carve((size_t)HIDDEN_ * DMODEL * 2);
  f16* s3h   = (f16*)carve((size_t)DMODEL * HIDDEN_ * 2);
  f16* Hs    = (f16*)carve((size_t)T_TOK * HIDDEN_ * 2);
  f16* Hb    = (f16*)carve((size_t)2 * T_TOK * RHID * 2);
  int* counts = (int*)carve(NE * 4);
  int* cursor = (int*)carve(NE * 4);
  int* offs   = (int*)carve(NE * 4);
  int* pe     = (int*)carve((size_t)T_TOK * 2 * 4);
  float* pg   = (float*)carve((size_t)T_TOK * 2 * 4);
  int* tokb   = (int*)carve((size_t)2 * T_TOK * 4);
  float* gb   = (float*)carve((size_t)2 * T_TOK * 4);

  auto cvt = [&](const float* s, f16* d, size_t n) {
    int n4 = (int)(n / 4);
    k_cvt<<<(n4 + 255) / 256, 256, 0, stream>>>((const float4*)s, (v4h*)d, n4);
  };
  cvt(x,   x16,  (size_t)T_TOK * DMODEL);
  cvt(w12, w12h, (size_t)NE * 2 * RHID * DMODEL);
  cvt(w3,  w3h,  (size_t)NE * DMODEL * RHID);
  cvt(sw1, s1h,  (size_t)HIDDEN_ * DMODEL);
  cvt(sw2, s2h,  (size_t)HIDDEN_ * DMODEL);
  cvt(sw3, s3h,  (size_t)DMODEL * HIDDEN_);

  // routing
  k_init<<<1, 32, 0, stream>>>(counts, cursor);
  k_router<<<T_TOK / 8, 256, 0, stream>>>(x, rw, bias, counts, pe, pg);
  k_offsets<<<1, 1, 0, stream>>>(counts, offs);
  k_scatter<<<T_TOK / 256, 256, 0, stream>>>(pe, pg, offs, cursor, tokb, gb);

  // shared expert
  k_shared_h<<<dim3(T_TOK / 64, HIDDEN_ / 128), 256, 0, stream>>>(x16, s1h, s2h, Hs);
  k_shared_out<<<dim3(T_TOK / 64, DMODEL / 128), 256, 0, stream>>>(Hs, s3h, out);

  // routed experts (dispatched top-2)
  k_routed_h<<<dim3(T_TOK / 64, RHID / 128, NE), 256, 0, stream>>>(
      x16, w12h, counts, offs, tokb, gb, Hb);
  k_routed_out<<<dim3(T_TOK / 64, DMODEL / 128, NE), 256, 0, stream>>>(
      Hb, w3h, counts, offs, tokb, out);
}
